// PhysicsGuidedTextureLoss_15169824489551
// MI455X (gfx1250) — compile-verified, matrix operand order checked by reading, not run on hardware
//
#include <hip/hip_runtime.h>
#include <math.h>

// ---------------- types / constants ----------------
typedef _Float16 f16;
typedef __attribute__((ext_vector_type(16))) _Float16 v16h;
typedef __attribute__((ext_vector_type(8)))  float    v8f;
typedef __attribute__((ext_vector_type(2)))  float    v2f;

#define NPIX   262144          // 512*512
#define NSIDE  512
#define EPSF   1e-6f
#define TWOPI  6.283185307179586f
#define RB     384             // padded radial bins (actual 362)
#define RBINS  362

// ---------------- workspace layout (bytes) ----------------
constexpr size_t ws_align(size_t x) { return (x + 255) & ~size_t(255); }
constexpr size_t OFF_MEANS  = 0;                                  // 16 f32
constexpr size_t OFF_LOSSES = ws_align(OFF_MEANS  + 64);          // 16 f32
constexpr size_t OFF_RCNT   = ws_align(OFF_LOSSES + 64);          // 384 f32
constexpr size_t OFF_RSUM   = ws_align(OFF_RCNT   + RB*4);        // 16*384 f32
constexpr size_t OFF_WIDTHS = ws_align(OFF_RSUM   + 16*RB*4);     // 32 f32
constexpr size_t OFF_QCOL   = ws_align(OFF_WIDTHS + 128);         // 16*512 f32
constexpr size_t OFF_QROW   = ws_align(OFF_QCOL   + 16*512*4);    // 16*512 f32
constexpr size_t OFF_CURVES = ws_align(OFF_QROW   + 16*512*4);    // 16*2*512 f32
constexpr size_t OFF_NKGM   = ws_align(OFF_CURVES + 16*1024*4);   // 16*961 f32
constexpr size_t OFF_NKGOM  = ws_align(OFF_NKGM   + 16*961*4);
constexpr size_t OFF_ALPHA  = ws_align(OFF_NKGOM  + 16*961*4);    // 16*512 f32
constexpr size_t OFF_G      = ws_align(OFF_ALPHA  + 16*512*4);    // 16*32*32 f32
constexpr size_t OFF_BK     = ws_align(OFF_G      + 16*1024*4);   // 256*32 f16
constexpr size_t OFF_C      = ws_align(OFF_BK     + 256*32*2);    // NPIX f32
constexpr size_t OFF_D      = ws_align(OFF_C      + NPIX*4);      // NPIX f32
constexpr size_t OFF_X      = ws_align(OFF_D      + NPIX*4);      // NPIX f32
constexpr size_t OFF_TR     = ws_align(OFF_X      + NPIX*4);      // NPIX f32
constexpr size_t OFF_TI     = ws_align(OFF_TR     + NPIX*4);      // NPIX f32
constexpr size_t OFF_PS     = ws_align(OFF_TI     + NPIX*4);      // NPIX f32
// total ~6.3 MB

// ---------------- WMMA helpers (ISA 7.12.2 layouts) ----------------
__device__ __forceinline__ v8f wmma_f16(v16h a, v16h b, v8f c) {
  return __builtin_amdgcn_wmma_f32_16x16x32_f16(false, a, false, b, (short)0, c,
                                                false, false);
}

__device__ __forceinline__ v8f wmma_f32(v2f a, v2f b, v8f c) {
  return __builtin_amdgcn_wmma_f32_16x16x4_f32(false, a, false, b, (short)0, c,
                                               false, false);
}

// f32 A fragment 16x4 at (row0,k0): lanes 0-15 K={0,1}, lanes 16-31 K={2,3}
__device__ __forceinline__ v2f fragA_f32(const float* A, int ld, int row0, int k0) {
  const int lane = threadIdx.x & 31;
  const int m = lane & 15, half = (lane >> 4) & 1;
  const float* p = A + (size_t)(row0 + m) * ld + k0 + 2 * half;
  v2f a;
  a[0] = p[0];
  a[1] = p[1];
  return a;
}

// f32 B fragment 4x16 at (k0,col0), row-major B[k][col]
__device__ __forceinline__ v2f fragB_f32(const float* B, int ld, int k0, int col0) {
  const int lane = threadIdx.x & 31;
  const int n = lane & 15, half = (lane >> 4) & 1;
  const float* p = B + (size_t)(k0 + 2 * half) * ld + col0 + n;
  v2f b;
  b[0] = p[0];
  b[1] = p[ld];
  return b;
}

// f16 B matrix fragment, row-major source B[k][col], 32x16 tile at (k0,col0)
__device__ __forceinline__ v16h fragB_rm(const f16* B, int ld, int k0, int col0) {
  const int lane = threadIdx.x & 31;
  const int n = lane & 15, half = (lane >> 4) & 1;
  v16h b;
#pragma unroll
  for (int v = 0; v < 8; ++v) {
    const int kb = ((v & 4) << 2) + half * 8 + ((v & 3) << 1);
    const f16* p = B + (size_t)(k0 + kb) * ld + (col0 + n);
    b[2 * v]     = p[0];
    b[2 * v + 1] = p[ld];
  }
  return b;
}

// store f32 accumulator tile (C/D layout) row-major
__device__ __forceinline__ void storeD_f32(float* Dst, int ld, int row0, int col0, v8f d) {
  const int lane = threadIdx.x & 31;
  const int n = lane & 15, half = (lane >> 4) & 1;
#pragma unroll
  for (int v = 0; v < 8; ++v)
    Dst[(size_t)(row0 + v + 8 * half) * ld + col0 + n] = d[v];
}

// ---------------- small kernels ----------------
__global__ void k_means(const float* __restrict__ ref, const float* __restrict__ pred,
                        float* __restrict__ means) {
  const int img = blockIdx.x;
  const float* src = (img < 8) ? (ref + (size_t)img * NPIX) : (pred + (size_t)(img - 8) * NPIX);
  __shared__ float red[256];
  float s = 0.f;
  for (int i = threadIdx.x; i < NPIX; i += 256) s += src[i];
  red[threadIdx.x] = s;
  __syncthreads();
  for (int st = 128; st > 0; st >>= 1) {
    if (threadIdx.x < st) red[threadIdx.x] += red[threadIdx.x + st];
    __syncthreads();
  }
  if (threadIdx.x == 0) means[img] = red[0] / (float)NPIX;
}

__global__ void k_gen_dft(float* __restrict__ C, float* __restrict__ D) {
  int idx = blockIdx.x * 256 + threadIdx.x;       // 1024 blocks
  int u = idx >> 9, k = idx & 511;
  float t = (float)((u * k) & 511) * (TWOPI / 512.f);
  C[idx] = cosf(t);
  D[idx] = -sinf(t);
}

__global__ void k_prep(const float* __restrict__ src, const float* __restrict__ means,
                       int img, float* __restrict__ X, int windowed) {
  int idx = blockIdx.x * 256 + threadIdx.x;
  int r = idx >> 9, c = idx & 511;
  float v = src[idx] - means[img];
  if (windowed) {
    float hr = 0.5f - 0.5f * cosf((float)r * (TWOPI / 512.f));
    float hc = 0.5f - 0.5f * cosf((float)c * (TWOPI / 512.f));
    v *= hr * hc;
  }
  X[idx] = v;
}

// ---------------- DFT-as-GEMM step A: Tr = C*X, Ti = D*X (f32 WMMA) ----------------
__global__ void __launch_bounds__(128) k_gemmA(const float* __restrict__ C,
                                               const float* __restrict__ D,
                                               const float* __restrict__ X,
                                               float* __restrict__ Tr,
                                               float* __restrict__ Ti) {
  const int wave = threadIdx.x >> 5;
  const int n0 = (blockIdx.x * 4 + wave) * 16;
  const int m0 = blockIdx.y * 16;
  v8f accR = {}, accI = {};
  for (int ks32 = 0; ks32 < 512; ks32 += 32) {
    if (ks32 + 32 < 512) __builtin_prefetch(X + (size_t)(ks32 + 32) * 512 + n0, 0, 1);
#pragma unroll
    for (int kk = 0; kk < 32; kk += 4) {
      const int ks = ks32 + kk;
      v2f aC = fragA_f32(C, 512, m0, ks);
      v2f aD = fragA_f32(D, 512, m0, ks);
      v2f bX = fragB_f32(X, 512, ks, n0);
      accR = wmma_f32(aC, bX, accR);
      accI = wmma_f32(aD, bX, accI);
    }
  }
  storeD_f32(Tr, 512, m0, n0, accR);
  storeD_f32(Ti, 512, m0, n0, accI);
}

// ---- step B: Fr = Tr*C - Ti*D, Fi = Tr*D + Ti*C ; emit power spectrum ----
__global__ void __launch_bounds__(128) k_gemmB(const float* __restrict__ Tr,
                                               const float* __restrict__ Ti,
                                               const float* __restrict__ C,
                                               const float* __restrict__ D,
                                               float* __restrict__ OutP, int logmode) {
  const int wave = threadIdx.x >> 5;
  const int n0 = (blockIdx.x * 4 + wave) * 16;
  const int m0 = blockIdx.y * 16;
  v8f aP = {}, aM = {}, aI = {};
  for (int ks32 = 0; ks32 < 512; ks32 += 32) {
    if (ks32 + 32 < 512) __builtin_prefetch(C + (size_t)(ks32 + 32) * 512 + n0, 0, 1);
#pragma unroll
    for (int kk = 0; kk < 32; kk += 4) {
      const int ks = ks32 + kk;
      v2f fR = fragA_f32(Tr, 512, m0, ks);
      v2f fI = fragA_f32(Ti, 512, m0, ks);
      v2f bC = fragB_f32(C, 512, ks, n0);
      v2f bD = fragB_f32(D, 512, ks, n0);
      aP = wmma_f32(fR, bC, aP);
      aM = wmma_f32(fI, bD, aM);
      aI = wmma_f32(fR, bD, aI);
      aI = wmma_f32(fI, bC, aI);
    }
  }
  const int lane = threadIdx.x & 31;
  const int n = lane & 15, half = (lane >> 4) & 1;
#pragma unroll
  for (int v = 0; v < 8; ++v) {
    float fr = aP[v] - aM[v];
    float fi = aI[v];
    float P = (fr * fr + fi * fi) * (1.f / 262144.f);   // ortho 1/512 squared
    float o = logmode ? logf(fmaxf(P, EPSF) + EPSF) : P;
    OutP[(size_t)(m0 + v + 8 * half) * 512 + n0 + n] = o;
  }
}

// ---------------- radial binning ----------------
__device__ __forceinline__ int radial_bin(int idx) {
  int u = idx >> 9, v = idx & 511;
  float dy = (float)u - 255.5f, dx = (float)v - 255.5f;
  return (int)rintf(sqrtf(dy * dy + dx * dx));   // matches np.round (half-even)
}

__global__ void k_rcnt(float* __restrict__ rcnt) {
  __shared__ float bins[RB];
  for (int i = threadIdx.x; i < RB; i += 256) bins[i] = 0.f;
  __syncthreads();
  int idx = blockIdx.x * 256 + threadIdx.x;
  atomicAdd(&bins[radial_bin(idx)], 1.f);
  __syncthreads();
  for (int i = threadIdx.x; i < RB; i += 256) atomicAdd(&rcnt[i], bins[i]);
}

__global__ void k_radial(const float* __restrict__ PS, float* __restrict__ rsumImg) {
  __shared__ float bins[RB];
  for (int i = threadIdx.x; i < RB; i += 256) bins[i] = 0.f;
  __syncthreads();
  int idx = blockIdx.x * 256 + threadIdx.x;
  atomicAdd(&bins[radial_bin(idx)], PS[idx]);
  __syncthreads();
  for (int i = threadIdx.x; i < RB; i += 256) atomicAdd(&rsumImg[i], bins[i]);
}

// ---------------- autocorr slice extraction ----------------
__global__ void k_extractQ(const float* __restrict__ P, float* __restrict__ qcol,
                           float* __restrict__ qrow) {
  int g = blockIdx.x * 256 + threadIdx.x;   // 4 blocks -> 1024 threads
  if (g < 512) {
    int u = g;
    float s = 0.f;
    for (int v = 0; v < 512; ++v) s += P[u * 512 + v] * ((v & 1) ? -1.f : 1.f);
    qcol[u] = s;
  } else {
    int v = g - 512;
    float s = 0.f;
    for (int u = 0; u < 512; ++u) s += P[u * 512 + v] * ((u & 1) ? -1.f : 1.f);
    qrow[v] = s;
  }
}

__global__ void k_curves(const float* __restrict__ qcol, const float* __restrict__ qrow,
                         float* __restrict__ curves) {
  int g = blockIdx.x * 256 + threadIdx.x;   // 64 blocks -> 16*2*512
  int img = g >> 10, rem = g & 1023, slice = rem >> 9, y = rem & 511;
  const float* q = (slice == 0 ? qcol : qrow) + img * 512;
  float s = 0.f;
  for (int u = 0; u < 512; ++u)
    s += q[u] * cosf((float)((u * y) & 511) * (TWOPI / 512.f));
  curves[g] = s;
}

__global__ void __launch_bounds__(512) k_hpw(const float* __restrict__ curves,
                                             float* __restrict__ widths) {
  __shared__ float c[512];
  __shared__ float rv[512];
  __shared__ int   ri[512];
  int t = threadIdx.x;
  c[t] = curves[blockIdx.x * 512 + t];
  rv[t] = c[t];
  __syncthreads();
  for (int s = 256; s > 0; s >>= 1) {
    if (t < s) rv[t] = fmaxf(rv[t], rv[t + s]);
    __syncthreads();
  }
  float peak = fmaxf(rv[0], EPSF);
  __syncthreads();
  rv[t] = fabsf(c[t] - 0.5f * peak);
  ri[t] = t;
  __syncthreads();
  for (int s = 256; s > 0; s >>= 1) {
    if (t < s) {
      if (rv[t + s] < rv[t] || (rv[t + s] == rv[t] && ri[t + s] < ri[t])) {
        rv[t] = rv[t + s];
        ri[t] = ri[t + s];
      }
    }
    __syncthreads();
  }
  if (t == 0) widths[blockIdx.x] = 2.f * (float)ri[0];
}

// ---------------- spectral losses (radial MSE + slope/intercept fits) ----------------
__global__ void __launch_bounds__(256) k_spec_losses(const float* __restrict__ rsum,
                                                     const float* __restrict__ rcnt,
                                                     float* __restrict__ losses) {
  __shared__ float ra[16][RB];
  for (int i = threadIdx.x; i < 16 * RB; i += 256) {
    int img = i / RB, r = i % RB;
    ra[img][r] = rsum[i] / (rcnt[r] + EPSF);
  }
  __syncthreads();
  if (threadIdx.x == 0) {
    float mean[16], sd[16];
    for (int img = 0; img < 16; ++img) {
      float m = 0.f;
      for (int r = 0; r < RBINS; ++r) m += ra[img][r];
      m /= (float)RBINS;
      float v = 0.f;
      for (int r = 0; r < RBINS; ++r) { float d = ra[img][r] - m; v += d * d; }
      mean[img] = m;
      sd[img] = sqrtf(v / (float)(RBINS - 1));   // ddof=1
    }
    float mse = 0.f;
    for (int b = 0; b < 8; ++b)
      for (int r = 0; r < RBINS; ++r) {
        float zr = (ra[b][r] - mean[b]) / (sd[b] + EPSF);
        float zp = (ra[8 + b][r] - mean[8 + b]) / (sd[8 + b] + EPSF);
        float d = zp - zr;
        mse += d * d;
      }
    mse /= (8.f * (float)RBINS);
    // slope fit on bins [36,181)
    float la = 0.f, lb = 0.f;
    for (int b = 0; b < 8; ++b) {
      float fit[2][2];
      for (int s = 0; s < 2; ++s) {
        int img = s * 8 + b;
        float sx = 0, sxx = 0, sy = 0, sxy = 0, n = 0;
        for (int r = 36; r < 181; ++r) {
          float x = (float)r, y = ra[img][r];
          sx += x; sxx += x * x; sy += y; sxy += x * y; n += 1.f;
        }
        float m00 = sxx + EPSF, m01 = sx, m11 = n + EPSF;
        float det = m00 * m11 - m01 * m01;
        fit[s][0] = (m11 * sxy - m01 * sy) / det;
        fit[s][1] = (-m01 * sxy + m00 * sy) / det;
      }
      la += fabsf(fit[1][0] - fit[0][0]);
      lb += fabsf(fit[1][1] - fit[0][1]);
    }
    losses[0] = 1.0f * mse;                       // W_RADIAL
    losses[1] = 0.2f * (la / 8.f) + 0.2f * (lb / 8.f);
  }
}

// ---------------- Nakagami moments ----------------
__global__ void k_nkg(const float* __restrict__ ref, const float* __restrict__ pred,
                      float* __restrict__ mOut, float* __restrict__ omOut) {
  const int img = blockIdx.z;
  const int widx = blockIdx.x;
  const int wy = widx / 31, wx = widx % 31;
  const float* src = (img < 8) ? (ref + (size_t)img * NPIX) : (pred + (size_t)(img - 8) * NPIX);
  const int h0 = wy * 16, w0 = wx * 16;
  float s2 = 0.f, s4 = 0.f;
  for (int i = threadIdx.x; i < 1024; i += 256) {
    int r = i >> 5, c = i & 31;
    float e = fmaxf(src[(h0 + r) * 512 + (w0 + c)], EPSF);
    float e2 = e * e;
    s2 += e2;
    s4 += e2 * e2;
  }
  __shared__ float r2[256], r4[256];
  r2[threadIdx.x] = s2; r4[threadIdx.x] = s4;
  __syncthreads();
  for (int s = 128; s > 0; s >>= 1) {
    if (threadIdx.x < s) { r2[threadIdx.x] += r2[threadIdx.x + s]; r4[threadIdx.x] += r4[threadIdx.x + s]; }
    __syncthreads();
  }
  if (threadIdx.x == 0) {
    float A2 = r2[0] / 1024.f, A4 = r4[0] / 1024.f;
    float var = fmaxf(A4 - A2 * A2, EPSF);
    mOut[img * 961 + widx] = fmaxf(A2 * A2 / var, EPSF);
    omOut[img * 961 + widx] = A2;
  }
}

// ---------------- axial attenuation fit ----------------
__global__ void k_alpha(const float* __restrict__ ref, const float* __restrict__ pred,
                        float* __restrict__ alpha) {
  int g = blockIdx.x * 256 + threadIdx.x;   // 32 blocks -> 8192
  int img = g >> 9, w = g & 511;
  const float* src = (img < 8) ? (ref + (size_t)img * NPIX) : (pred + (size_t)(img - 8) * NPIX);
  float Sx = 0.f, Sxx = 0.f;
  for (int z = 102; z < 409; ++z) { Sx += (float)z; Sxx += (float)z * (float)z; }
  float n = 307.f;
  float m00 = Sxx + EPSF, m01 = Sx, m11 = n + EPSF;
  float det = m00 * m11 - m01 * m01;
  float i00 = m11 / det, i01 = -m01 / det;
  float s = 0.f;
  for (int z = 102; z < 409; ++z)
    s += (i00 * (float)z + i01) * logf(fmaxf(src[z * 512 + w], EPSF));
  alpha[g] = -s;
}

// ---------------- Gabor conv (implicit GEMM) + Gram (f16 WMMA) ----------------
__global__ void k_prep_gabor(const float* __restrict__ kern, f16* __restrict__ Bk) {
  int idx = blockIdx.x * 256 + threadIdx.x;   // 32 blocks -> 8192 = 256*32
  int k = idx >> 5, n = idx & 31;
  float v = (k < 225 && n < 18) ? kern[n * 225 + k] : 0.f;
  Bk[idx] = (f16)v;
}

__device__ __forceinline__ v16h conv_fragA(const f16* patch, int m0, int k0) {
  const int lane = threadIdx.x & 31;
  const int mloc = lane & 15, half = (lane >> 4) & 1;
  const int m = m0 + mloc;
  const int py = m >> 4, px = m & 15;
  v16h a;
#pragma unroll
  for (int v = 0; v < 8; ++v) {
    const int kb = ((v & 4) << 2) + half * 8 + ((v & 3) << 1);
#pragma unroll
    for (int e = 0; e < 2; ++e) {
      int k = k0 + kb + e;
      f16 val = (f16)0.f;
      if (k < 225) {
        int dy = k / 15, dx = k % 15;
        val = patch[(py + dy) * 32 + (px + dx)];
      }
      a[2 * v + e] = val;
    }
  }
  return a;
}

__device__ __forceinline__ v16h fragAT_lds(const f16* F, int ch0, int k0) {
  const int lane = threadIdx.x & 31;
  const int m = ch0 + (lane & 15), half = (lane >> 4) & 1;
  v16h a;
#pragma unroll
  for (int v = 0; v < 8; ++v) {
    const int kb = ((v & 4) << 2) + half * 8 + ((v & 3) << 1);
    a[2 * v]     = F[(k0 + kb) * 32 + m];
    a[2 * v + 1] = F[(k0 + kb + 1) * 32 + m];
  }
  return a;
}

__global__ void __launch_bounds__(128) k_gabor(const float* __restrict__ ref,
                                               const float* __restrict__ pred,
                                               const f16* __restrict__ Bk,
                                               float* __restrict__ G) {
  const int img = blockIdx.z;
  const float* src = (img < 8) ? (ref + (size_t)img * NPIX) : (pred + (size_t)(img - 8) * NPIX);
  const int x0 = blockIdx.x * 16;
  const int tid = threadIdx.x, wave = tid >> 5, lane = tid & 31;
  __shared__ f16 patch[30 * 32];
  __shared__ f16 Flds[256 * 32];
  const int mi = wave >> 1, ni = wave & 1;   // gram tile per wave
  v8f accG = {};
  for (int ty = 0; ty < 32; ++ty) {
    const int y0 = ty * 16;
    __syncthreads();
    // stage 30x30 input patch (zero pad 7 for SAME conv)
    for (int i = tid; i < 30 * 32; i += 128) {
      int r = i >> 5, c = i & 31;
      float v = 0.f;
      if (c < 30) {
        int gy = y0 - 7 + r, gx = x0 - 7 + c;
        if (gy >= 0 && gy < 512 && gx >= 0 && gx < 512) v = src[gy * 512 + gx];
      }
      patch[i] = (f16)v;
    }
    __syncthreads();
    // implicit-GEMM conv: M=256 px, N=32 (18 padded), K=256 (225 padded)
    v8f acc[4][2];
#pragma unroll
    for (int j = 0; j < 4; ++j) { acc[j][0] = (v8f){}; acc[j][1] = (v8f){}; }
    for (int ks = 0; ks < 256; ks += 32) {
      v16h b0 = fragB_rm(Bk, 32, ks, 0);
      v16h b1 = fragB_rm(Bk, 32, ks, 16);
#pragma unroll
      for (int j = 0; j < 4; ++j) {
        v16h a = conv_fragA(patch, (wave * 4 + j) * 16, ks);
        acc[j][0] = wmma_f16(a, b0, acc[j][0]);
        acc[j][1] = wmma_f16(a, b1, acc[j][1]);
      }
    }
    // write feature tile F [256 px][32 ch] to LDS as f16
    {
      const int n = lane & 15, half = (lane >> 4) & 1;
#pragma unroll
      for (int j = 0; j < 4; ++j)
#pragma unroll
        for (int nt = 0; nt < 2; ++nt)
#pragma unroll
          for (int v = 0; v < 8; ++v)
            Flds[((wave * 4 + j) * 16 + v + 8 * half) * 32 + nt * 16 + n] =
                (f16)acc[j][nt][v];
    }
    __syncthreads();
    // Gram accumulate: G += F^T * F  (M=N=32 ch, K=256 px)
    for (int ks = 0; ks < 256; ks += 32) {
      v16h aT = fragAT_lds(Flds, mi * 16, ks);
      v16h bF = fragB_rm(Flds, 32, ks, ni * 16);
      accG = wmma_f16(aT, bF, accG);
    }
  }
  const int n = lane & 15, half = (lane >> 4) & 1;
#pragma unroll
  for (int v = 0; v < 8; ++v)
    atomicAdd(&G[img * 1024 + (mi * 16 + v + 8 * half) * 32 + ni * 16 + n], accG[v]);
}

// ---------------- final combine ----------------
__global__ void __launch_bounds__(256) k_final(const float* __restrict__ nkgm,
                                               const float* __restrict__ nkgom,
                                               const float* __restrict__ alpha,
                                               const float* __restrict__ widths,
                                               const float* __restrict__ G,
                                               const float* __restrict__ losses,
                                               float* __restrict__ out) {
  int tid = threadIdx.x;
  float sm = 0.f, som = 0.f, sal = 0.f, sg = 0.f;
  for (int i = tid; i < 8 * 961; i += 256) {
    int b = i / 961, w = i % 961;
    sm  += fabsf(nkgm[(8 + b) * 961 + w] - nkgm[b * 961 + w]);
    som += fabsf(nkgom[(8 + b) * 961 + w] - nkgom[b * 961 + w]);
  }
  for (int i = tid; i < 8 * 512; i += 256)
    sal += fabsf(alpha[i + 8 * 512] - alpha[i]);
  for (int i = tid; i < 8 * 324; i += 256) {
    int b = i / 324, r = i % 324;
    int n = r / 18, m = r % 18;
    float gr = G[b * 1024 + n * 32 + m] * (1.f / 262144.f);
    float gp = G[(8 + b) * 1024 + n * 32 + m] * (1.f / 262144.f);
    float d = gp - gr;
    sg += d * d;
  }
  __shared__ float R0[256], R1[256], R2[256], R3[256];
  R0[tid] = sm; R1[tid] = som; R2[tid] = sal; R3[tid] = sg;
  __syncthreads();
  for (int s = 128; s > 0; s >>= 1) {
    if (tid < s) {
      R0[tid] += R0[tid + s]; R1[tid] += R1[tid + s];
      R2[tid] += R2[tid + s]; R3[tid] += R3[tid + s];
    }
    __syncthreads();
  }
  if (tid == 0) {
    float loss_nkg   = 0.5f * (R0[0] / (8.f * 961.f)) + 0.25f * (R1[0] / (8.f * 961.f));
    float loss_alpha = 0.2f * (R2[0] / (8.f * 512.f));
    float loss_gram  = 0.2f * (R3[0] / (8.f * 324.f));
    float wa = 0.f, wl = 0.f;
    for (int b = 0; b < 8; ++b) {
      wa += fabsf(widths[(8 + b) * 2 + 0] - widths[b * 2 + 0]);
      wl += fabsf(widths[(8 + b) * 2 + 1] - widths[b * 2 + 1]);
    }
    float loss_auto = 0.5f * (wa / 8.f) + 0.5f * (wl / 8.f);
    out[0] = losses[0] + losses[1] + loss_auto + loss_nkg + loss_alpha + loss_gram;
  }
}

// ---------------- host driver ----------------
extern "C" void kernel_launch(void* const* d_in, const int* in_sizes, int n_in,
                              void* d_out, int out_size, void* d_ws, size_t ws_size,
                              hipStream_t stream) {
  (void)in_sizes; (void)n_in; (void)out_size; (void)ws_size;
  const float* Iref  = (const float*)d_in[0];
  const float* Ipred = (const float*)d_in[1];
  const float* kern  = (const float*)d_in[2];
  float* out = (float*)d_out;
  char* ws = (char*)d_ws;

  float* means  = (float*)(ws + OFF_MEANS);
  float* losses = (float*)(ws + OFF_LOSSES);
  float* rcnt   = (float*)(ws + OFF_RCNT);
  float* rsum   = (float*)(ws + OFF_RSUM);
  float* widths = (float*)(ws + OFF_WIDTHS);
  float* qcol   = (float*)(ws + OFF_QCOL);
  float* qrow   = (float*)(ws + OFF_QROW);
  float* curves = (float*)(ws + OFF_CURVES);
  float* nkgm   = (float*)(ws + OFF_NKGM);
  float* nkgom  = (float*)(ws + OFF_NKGOM);
  float* alpha  = (float*)(ws + OFF_ALPHA);
  float* G      = (float*)(ws + OFF_G);
  f16*   Bk     = (f16*)(ws + OFF_BK);
  float* Cm     = (float*)(ws + OFF_C);
  float* Dm     = (float*)(ws + OFF_D);
  float* X      = (float*)(ws + OFF_X);
  float* Tr     = (float*)(ws + OFF_TR);
  float* Ti     = (float*)(ws + OFF_TI);
  float* PS     = (float*)(ws + OFF_PS);

  hipMemsetAsync(rsum, 0, 16 * RB * 4, stream);
  hipMemsetAsync(rcnt, 0, RB * 4, stream);
  hipMemsetAsync(G, 0, 16 * 1024 * 4, stream);
  hipMemsetAsync(losses, 0, 64, stream);

  k_means<<<16, 256, 0, stream>>>(Iref, Ipred, means);
  k_gen_dft<<<NPIX / 256, 256, 0, stream>>>(Cm, Dm);
  k_rcnt<<<NPIX / 256, 256, 0, stream>>>(rcnt);

  for (int img = 0; img < 16; ++img) {
    const float* src = (img < 8) ? (Iref + (size_t)img * NPIX)
                                 : (Ipred + (size_t)(img - 8) * NPIX);
    // windowed pipeline -> log power spectrum -> radial sums
    k_prep<<<NPIX / 256, 256, 0, stream>>>(src, means, img, X, 1);
    k_gemmA<<<dim3(8, 32), 128, 0, stream>>>(Cm, Dm, X, Tr, Ti);
    k_gemmB<<<dim3(8, 32), 128, 0, stream>>>(Tr, Ti, Cm, Dm, PS, 1);
    k_radial<<<NPIX / 256, 256, 0, stream>>>(PS, rsum + img * RB);
    // plain pipeline -> power spectrum -> autocorr slice weights
    k_prep<<<NPIX / 256, 256, 0, stream>>>(src, means, img, X, 0);
    k_gemmA<<<dim3(8, 32), 128, 0, stream>>>(Cm, Dm, X, Tr, Ti);
    k_gemmB<<<dim3(8, 32), 128, 0, stream>>>(Tr, Ti, Cm, Dm, PS, 0);
    k_extractQ<<<4, 256, 0, stream>>>(PS, qcol + img * 512, qrow + img * 512);
  }

  k_spec_losses<<<1, 256, 0, stream>>>(rsum, rcnt, losses);
  k_curves<<<64, 256, 0, stream>>>(qcol, qrow, curves);
  k_hpw<<<32, 512, 0, stream>>>(curves, widths);
  k_nkg<<<dim3(961, 1, 16), 256, 0, stream>>>(Iref, Ipred, nkgm, nkgom);
  k_alpha<<<32, 256, 0, stream>>>(Iref, Ipred, alpha);
  k_prep_gabor<<<32, 256, 0, stream>>>(kern, Bk);
  k_gabor<<<dim3(32, 1, 16), 128, 0, stream>>>(Iref, Ipred, Bk, G);
  k_final<<<1, 256, 0, stream>>>(nkgm, nkgom, alpha, widths, G, losses, out);
}
